// Model_GNN_paf_29454885716672
// MI455X (gfx1250) — matrix operand order repaired and checked by reading, not compiled, not run on hardware
//
#include <hip/hip_runtime.h>

// ---------------------------------------------------------------------------
// Types for CDNA5 WMMA (wave32): A,B = 16 bf16 (8 VGPRs), C/D = 8 f32.
// ---------------------------------------------------------------------------
typedef __attribute__((ext_vector_type(16))) __bf16 v16bf;
typedef __attribute__((ext_vector_type(8)))  float  v8f;
typedef __attribute__((ext_vector_type(4)))  unsigned int u32x4;
typedef __attribute__((ext_vector_type(8)))  unsigned int u32x8;

union Frag {
    unsigned int u[8];
    v16bf        v;
};

// Padded activation geometry: every stage uses [plane][132][132][32] bf16,
// with a 2-pixel zero border so 5x5 halo loads never leave the buffer.
#define PADW   132
#define CP     32
#define PLANE  (PADW * PADW * CP)          // 557,568 elements per plane

__device__ __forceinline__ unsigned short f2bf(float f) {
    unsigned int u = __builtin_bit_cast(unsigned int, f);
    u += 0x7FFFu + ((u >> 16) & 1u);   // round-to-nearest-even
    return (unsigned short)(u >> 16);
}

// ---------------------------------------------------------------------------
// Kernel 1: channel gather (57 -> 19x4) + f32->bf16 into padded layout.
// ---------------------------------------------------------------------------
__global__ __launch_bounds__(256) void gather_bf16(
    const float* __restrict__ cnn,        // [16][57][128][128]
    unsigned short* __restrict__ s0)      // [304][132][132][32] bf16
{
    static const int CHT[19][4] = {
        { 1,  8, 19, 20}, { 8,  9, 21, 22}, { 9, 10, 23, 24}, { 1, 11, 25, 26},
        {11, 12, 27, 28}, {12, 13, 29, 30}, { 1,  2, 31, 32}, { 2,  3, 33, 34},
        { 3,  4, 35, 36}, { 2, 16, 37, 38}, { 1,  5, 39, 40}, { 5,  6, 41, 42},
        { 6,  7, 43, 44}, { 5, 17, 45, 46}, { 0,  1, 47, 48}, { 0, 14, 49, 50},
        { 0, 15, 51, 52}, {14, 16, 53, 54}, {15, 17, 55, 56}
    };
    long long p = (long long)blockIdx.x * 256 + threadIdx.x;   // exact grid
    int chunk = (int)(p >> 14);       // img*19 + e
    int img   = chunk / 19;
    int e     = chunk % 19;
    int yx    = (int)(p & 16383);
    int y     = yx >> 7;
    int x     = yx & 127;
    const float* base = cnn + (long long)img * 57 * 16384 + yx;
    unsigned long long pack = 0;
#pragma unroll
    for (int c = 0; c < 4; ++c) {
        unsigned short h = f2bf(base[(long long)CHT[e][c] * 16384]);
        pack |= (unsigned long long)h << (16 * c);
    }
    unsigned short* dst = s0 + (long long)chunk * PLANE
                             + ((long long)(y + 2) * PADW + (x + 2)) * CP;
    *(unsigned long long*)dst = pack;   // 64-byte aligned
}

// ---------------------------------------------------------------------------
// Kernel 2: weight repack f32 (O,I,KH,KW) -> bf16 [O][KK][32] (cin padded 32).
// ---------------------------------------------------------------------------
__global__ __launch_bounds__(256) void wconv(
    const float* __restrict__ src, unsigned short* __restrict__ dst,
    int O, int I, int KK, int total)
{
    int t = blockIdx.x * 256 + threadIdx.x;
    if (t >= total) return;
    int ip  = t & 31;
    int r   = t >> 5;
    int tap = r % KK;
    int o   = r / KK;
    float v = (ip < I) ? src[(o * I + ip) * KK + tap] : 0.0f;
    dst[t] = f2bf(v);
}

// ---------------------------------------------------------------------------
// Kernel 3: implicit-GEMM 5x5 conv, bf16 channel-last, f32 accum via WMMA.
// Block: 256 thr = 8 waves. Tile: 4 output rows x 64 cols of one plane.
// LDS staging via TDM: wave w DMAs halo row w (68x32 bf16 = 4352B) with
// tensor_load_to_lds, then s_wait_tensorcnt + barrier.
// ---------------------------------------------------------------------------
template<int CPIN, int COUT>
__global__ __launch_bounds__(256) void conv5x5(
    const unsigned short* __restrict__ in,    // [304][132][132][32] bf16
    unsigned short*       __restrict__ out,   // [304][132][132][32] bf16
    const unsigned short* __restrict__ wts,   // [COUT][25][32] bf16
    const float*          __restrict__ bias)  // [COUT] f32
{
    constexpr int TY = 4, TX = 64, LR = TY + 4, LW = TX + 4;   // 8 x 68 tile
    constexpr int MT = COUT / 16;
    __shared__ __align__(128) unsigned short lds[LR * LW * CP];  // 34,816 B

    const int bx  = blockIdx.x;        // 0..1   (x segment of 64)
    const int by  = blockIdx.y;        // 0..31  (row band of 4)
    const int img = blockIdx.z;        // 0..303 (plane)
    const int tid  = threadIdx.x;
    const int lane = tid & 31;
    const int wv   = tid >> 5;         // wave id, uniform per wave
    const int half = lane >> 4;
    const int ln16 = lane & 15;

    // ---- TDM: wave wv stages halo row wv of the tile into LDS ----
    {
        const unsigned short* gsrc = in + (long long)img * PLANE
            + ((long long)(by * TY + wv) * PADW + bx * TX) * CP;  // padded coords
        unsigned long long gaddr = (unsigned long long)(uintptr_t)gsrc;
        unsigned lds_off = (unsigned)(unsigned long long)(uintptr_t)&lds[wv * LW * CP];
        const unsigned td0 = 0x40000000u;   // tensor_dim0 (just >= tile span)
        const unsigned td1 = 16u;           // tensor_dim1 (>= 1 row)
        u32x4 g0;
        g0[0] = 1u;                                   // count=1, flags=0
        g0[1] = lds_off;                              // lds_addr
        g0[2] = (unsigned)gaddr;                      // global_addr[31:0]
        g0[3] = ((unsigned)(gaddr >> 32) & 0x01FFFFFFu) | 0x80000000u; // [56:32] | type=2
        u32x8 g1;
        g1[0] = 0x00010000u;                          // data_size=1 (2 bytes)
        g1[1] = (td0 & 0xFFFFu) << 16;                // tensor_dim0 lo
        g1[2] = ((td0 >> 16) & 0xFFFFu) | ((td1 & 0xFFFFu) << 16);
        g1[3] = ((td1 >> 16) & 0xFFFFu) | ((unsigned)(LW * CP) << 16); // tile_dim0
        g1[4] = 1u;                                   // tile_dim1=1, tile_dim2=0
        g1[5] = (unsigned)(PADW * CP);                // tensor_dim0_stride lo
        g1[6] = 0u;                                   // stride0 hi | stride1 lo
        g1[7] = 0u;
        asm volatile("tensor_load_to_lds %0, %1" :: "s"(g0), "s"(g1) : "memory");
        __builtin_amdgcn_s_wait_tensorcnt(0);
    }
    __syncthreads();

    const int ry   = wv >> 1;          // local output row 0..3
    const int xseg = (wv & 1) * 32;    // local x base 0 or 32

    v8f acc[MT][2];
#pragma unroll
    for (int mt = 0; mt < MT; ++mt) {
        acc[mt][0] = {};
        acc[mt][1] = {};
    }

    const unsigned short* wrow[MT];
#pragma unroll
    for (int mt = 0; mt < MT; ++mt)
        wrow[mt] = wts + (size_t)(mt * 16 + ln16) * 25 * 32;   // A row m = lane&15

    for (int ky = 0; ky < 5; ++ky) {
        const int row = ry + ky;                                // 0..7 in LDS
#pragma unroll
        for (int kx = 0; kx < 5; ++kx) {
            const int tap = ky * 5 + kx;
            // ---- B fragments (32x16 bf16): k = cin, n = pixel x ----
            Frag B0, B1;
            const int x0 = xseg + ln16 + kx;                    // 0..67
#pragma unroll
            for (int v = 0; v < 8; ++v) {
                int k0 = 16 * half + 2 * v;
                if (k0 < CPIN) {
                    B0.u[v] = *(const unsigned int*)&lds[(row * LW + x0) * CP + k0];
                    B1.u[v] = *(const unsigned int*)&lds[(row * LW + x0 + 16) * CP + k0];
                } else {
                    B0.u[v] = 0;
                    B1.u[v] = 0;
                }
            }
            // ---- A fragments (16x32 bf16): m = cout, k = cin ----
#pragma unroll
            for (int mt = 0; mt < MT; ++mt) {
                Frag A;
#pragma unroll
                for (int v = 0; v < 8; ++v) {
                    int k0 = 16 * half + 2 * v;
                    A.u[v] = *(const unsigned int*)&wrow[mt][tap * 32 + k0];
                }
                acc[mt][0] = __builtin_amdgcn_wmma_f32_16x16x32_bf16(
                    false, A.v, false, B0.v, (short)0, acc[mt][0], false, false);
                acc[mt][1] = __builtin_amdgcn_wmma_f32_16x16x32_bf16(
                    false, A.v, false, B1.v, (short)0, acc[mt][1], false, false);
            }
        }
    }

    // ---- epilogue: bias + ReLU + bf16 store into padded interior ----
    const int gy = by * TY + ry;
    const long long obase = (long long)img * PLANE;
#pragma unroll
    for (int mt = 0; mt < MT; ++mt) {
#pragma unroll
        for (int nt = 0; nt < 2; ++nt) {
            const int gx = bx * TX + xseg + nt * 16 + ln16;     // C tile: n = lane&15
            unsigned short* op = out + obase
                + ((long long)(gy + 2) * PADW + (gx + 2)) * CP;
#pragma unroll
            for (int v = 0; v < 8; ++v) {
                int co = mt * 16 + v + 8 * half;                // C tile: m = v + 8*half
                float r = acc[mt][nt][v] + bias[co];
                r = fmaxf(r, 0.0f);
                op[co] = f2bf(r);
            }
        }
    }
}

// ---------------------------------------------------------------------------
// Kernel 4: fused 1x1 convs: 32 -> 128 (ReLU, WMMA) -> 2 (f32 dot), f32 out.
// One wave per 16-pixel N-tile; hidden stays in registers.
// ---------------------------------------------------------------------------
__global__ __launch_bounds__(256) void fused1x1(
    const unsigned short* __restrict__ in,   // [304][132][132][32] bf16
    float*                __restrict__ outp, // [304*2*16384] f32
    const unsigned short* __restrict__ w5,   // [128][32] bf16
    const float*          __restrict__ b5,   // [128]
    const float*          __restrict__ w6,   // [2][128] f32
    const float*          __restrict__ b6)   // [2]
{
    const int lane = threadIdx.x & 31;
    const int half = lane >> 4;
    const int ln16 = lane & 15;
    const int wid  = blockIdx.x * 8 + (threadIdx.x >> 5);
    const int plane = wid >> 10;             // / (128 rows * 8 xtiles)
    const int rem   = wid & 1023;
    const int y     = rem >> 3;
    const int xt    = rem & 7;

    // B fragment: k = cin (32), n = pixel (16 consecutive x)
    const unsigned short* ib = in + (long long)plane * PLANE
        + ((long long)(y + 2) * PADW + (xt * 16 + ln16 + 2)) * CP;
    Frag B;
#pragma unroll
    for (int v = 0; v < 8; ++v)
        B.u[v] = *(const unsigned int*)&ib[16 * half + 2 * v];

    float q0 = 0.0f, q1 = 0.0f;
#pragma unroll
    for (int mt = 0; mt < 8; ++mt) {        // 128 hidden = 8 M-tiles
        Frag A;
#pragma unroll
        for (int v = 0; v < 8; ++v) {
            int k0 = 16 * half + 2 * v;
            A.u[v] = *(const unsigned int*)&w5[(size_t)(mt * 16 + ln16) * 32 + k0];
        }
        v8f c = {};
        c = __builtin_amdgcn_wmma_f32_16x16x32_bf16(
            false, A.v, false, B.v, (short)0, c, false, false);
#pragma unroll
        for (int v = 0; v < 8; ++v) {
            int h = mt * 16 + v + 8 * half;          // C tile row m
            float hv = fmaxf(c[v] + b5[h], 0.0f);    // ReLU(conv5)
            q0 += hv * w6[h];
            q1 += hv * w6[128 + h];
        }
    }
    // combine the two lane-halves that own complementary hidden rows
    q0 += __shfl_xor(q0, 16, 32);
    q1 += __shfl_xor(q1, 16, 32);
    if (half == 0) {
        long long yx = (long long)y * 128 + xt * 16 + ln16;
        outp[((long long)plane * 2 + 0) * 16384 + yx] = q0 + b6[0];
        outp[((long long)plane * 2 + 1) * 16384 + yx] = q1 + b6[1];
    }
}

// ---------------------------------------------------------------------------
// Host-side orchestration.
// ---------------------------------------------------------------------------
extern "C" void kernel_launch(void* const* d_in, const int* in_sizes, int n_in,
                              void* d_out, int out_size, void* d_ws, size_t ws_size,
                              hipStream_t stream)
{
    (void)in_sizes; (void)n_in; (void)out_size; (void)ws_size;

    const float* cnn = (const float*)d_in[0];
    const float* w[7];
    const float* b[7];
    for (int i = 0; i < 7; ++i) {
        w[i] = (const float*)d_in[2 + 2 * i];
        b[i] = (const float*)d_in[3 + 2 * i];
    }

    const long long PIX      = 304LL * 128 * 128;       // interior pixels
    const long long BUFELEMS = 304LL * PLANE;           // padded plane buffer
    unsigned short* ws   = (unsigned short*)d_ws;
    unsigned short* bufA = ws;
    unsigned short* bufB = ws + BUFELEMS;
    unsigned short* wbuf = ws + 2 * BUFELEMS;

    // Zero both activation buffers (establishes the 2-pixel zero borders).
    hipMemsetAsync(d_ws, 0, (size_t)(2 * BUFELEMS) * sizeof(unsigned short), stream);

    static const int Osz[6] = {16, 16, 32, 32, 32, 128};
    static const int Isz[6] = { 4, 16, 16, 32, 32,  32};
    static const int Ksz[6] = {25, 25, 25, 25, 25,   1};
    unsigned short* wb[6];
    {
        size_t off = 0;
        for (int i = 0; i < 6; ++i) {
            wb[i] = wbuf + off;
            off += (size_t)Osz[i] * Ksz[i] * 32;
        }
    }

    // 1) repack weights to bf16 [O][tap][32]
    for (int i = 0; i < 6; ++i) {
        int total = Osz[i] * Ksz[i] * 32;
        wconv<<<(total + 255) / 256, 256, 0, stream>>>(w[i], wb[i], Osz[i], Isz[i], Ksz[i], total);
    }

    // 2) gather + convert input -> bufA interior (cin=4 of 32)
    gather_bf16<<<(int)(PIX / 256), 256, 0, stream>>>(cnn, bufA);

    // 3) five 5x5 conv layers, ping-pong bufA/bufB
    dim3 cgrid(2, 32, 304);
    conv5x5< 4, 16><<<cgrid, 256, 0, stream>>>(bufA, bufB, wb[0], b[0]);
    conv5x5<16, 16><<<cgrid, 256, 0, stream>>>(bufB, bufA, wb[1], b[1]);
    conv5x5<16, 32><<<cgrid, 256, 0, stream>>>(bufA, bufB, wb[2], b[2]);
    conv5x5<32, 32><<<cgrid, 256, 0, stream>>>(bufB, bufA, wb[3], b[3]);
    conv5x5<32, 32><<<cgrid, 256, 0, stream>>>(bufA, bufB, wb[4], b[4]);

    // 4) fused 1x1 convs -> f32 output
    fused1x1<<<(int)(PIX / 16 / 8), 256, 0, stream>>>(bufB, (float*)d_out,
                                                      wb[5], b[5], w[6], b[6]);
}
